// CoMet_71760313582021
// MI455X (gfx1250) — matrix operand order repaired and checked by reading, not compiled, not run on hardware
//
#include <hip/hip_runtime.h>
#include <hip/hip_bf16.h>

#define NST 16
#define NCOM 4
#define NH 256
#define BATCH 262144
#define BM 32
#define THREADS 256

typedef __attribute__((ext_vector_type(16))) _Float16 v16h;
typedef __attribute__((ext_vector_type(8)))  _Float16 v8h;
typedef __attribute__((ext_vector_type(8)))  float    v8f;

// ---- workspace layout (units: halves) ----
#define OFF_W1H 0                       // 256 x 32  (K padded 16->32), row n = W1[n][k]
#define OFF_W1T (OFF_W1H + 256*32)      // 16 x 256, row s = W1[:,s]
#define OFF_W2H (OFF_W1T + 16*256)      // 256 x 256, row n = W2[n][k]   (fwd)
#define OFF_W2T (OFF_W2H + 256*256)     // 256 x 256, row n = W2[:,n]    (bwd)
#define OFF_W3H (OFF_W2T + 256*256)
#define OFF_W3T (OFF_W3H + 256*256)
#define OFF_W4H (OFF_W3T + 256*256)     // 20 x 256
#define WS_HALVES (OFF_W4H + 20*256)

__device__ __forceinline__ v8f wmma_f16(v16h a, v16h b, v8f c) {
  return __builtin_amdgcn_wmma_f32_16x16x32_f16(false, a, false, b, (short)0, c,
                                                false, false);
}

// A-fragment (16x32 f16) from LDS, row-major, stride lda halves.
// lane L holds row m; halves 0..7 = K base..base+7, halves 8..15 = K base+16..base+23,
// base = 8 when L>=16 else 0.
__device__ __forceinline__ v16h frag_a_lds(const _Float16* base, int lda, int m,
                                           int k0, int lane) {
  int koff = k0 + ((lane & 16) ? 8 : 0);
  const _Float16* p = base + m * lda + koff;
  v8h lo = *(const v8h*)p;
  v8h hi = *(const v8h*)(p + 16);
  v16h r;
#pragma unroll
  for (int i = 0; i < 8; ++i) { r[i] = lo[i]; r[i + 8] = hi[i]; }
  return r;
}

// B-fragment (32x16 f16) from global; weights stored as N x K rows (stride ldk halves).
// lane L holds column n = L&15; halves 0..7 = K base..base+7, halves 8..15 = K base+16..base+23,
// base = 8 when L>=16 else 0.
__device__ __forceinline__ v16h frag_b_gbl(const _Float16* base, int ldk, int n0,
                                           int k0, int lane) {
  int n = n0 + (lane & 15);
  int koff = k0 + ((lane & 16) ? 8 : 0);
  const _Float16* p = base + n * ldk + koff;
  v8h lo = *(const v8h*)p;
  v8h hi = *(const v8h*)(p + 16);
  v16h r;
#pragma unroll
  for (int i = 0; i < 8; ++i) { r[i] = lo[i]; r[i + 8] = hi[i]; }
  return r;
}

// Fast fused activation pair using HW transcendentals:
//   ls = log_sigmoid(u) = min(u,0) - log(1 + exp(-|u|))     (v_exp + v_log)
//   sn = sigmoid(-u)    = u>=0 ? e/(1+e) : 1/(1+e),  e = exp(-|u|)   (v_rcp)
__device__ __forceinline__ void act_pair(float u, float& ls, float& sn) {
  float e = __expf(-fabsf(u));
  float r = __builtin_amdgcn_rcpf(1.f + e);
  sn = (u >= 0.f) ? e * r : r;
  ls = fminf(u, 0.f) - __logf(1.f + e);
}

// ---------------------------------------------------------------------------
// Kernel 1: convert f32 weights to f16 layouts in workspace
// ---------------------------------------------------------------------------
__global__ void CoMet_convert_weights(const float* __restrict__ W1,
                                      const float* __restrict__ W2,
                                      const float* __restrict__ W3,
                                      const float* __restrict__ W4,
                                      _Float16* __restrict__ ws) {
  _Float16* W1h = ws + OFF_W1H;
  _Float16* W1t = ws + OFF_W1T;
  _Float16* W2h = ws + OFF_W2H;
  _Float16* W2t = ws + OFF_W2T;
  _Float16* W3h = ws + OFF_W3H;
  _Float16* W3t = ws + OFF_W3T;
  _Float16* W4h = ws + OFF_W4H;
  int tid = blockIdx.x * blockDim.x + threadIdx.x;
  int np = gridDim.x * blockDim.x;
  for (int i = tid; i < 256 * 32; i += np) {
    int n = i >> 5, k = i & 31;
    W1h[i] = (k < NST) ? (_Float16)W1[n * NST + k] : (_Float16)0.f;
  }
  for (int i = tid; i < NST * NH; i += np) {
    int s = i >> 8, h = i & 255;
    W1t[i] = (_Float16)W1[h * NST + s];
  }
  for (int i = tid; i < NH * NH; i += np) {
    int n = i >> 8, k = i & 255;
    W2h[i] = (_Float16)W2[i];
    W2t[i] = (_Float16)W2[k * NH + n];
    W3h[i] = (_Float16)W3[i];
    W3t[i] = (_Float16)W3[k * NH + n];
  }
  for (int i = tid; i < (NST + NCOM) * NH; i += np) W4h[i] = (_Float16)W4[i];
}

// ---------------------------------------------------------------------------
// Kernel 2: fused forward + 4-seed backward Jacobian + orthogonal projection
// ---------------------------------------------------------------------------
__global__ __launch_bounds__(THREADS) void CoMet_fused(
    const float* __restrict__ x, const _Float16* __restrict__ ws,
    float* __restrict__ out) {
  const _Float16* W1h = ws + OFF_W1H;
  const _Float16* W1t = ws + OFF_W1T;
  const _Float16* W2h = ws + OFF_W2H;
  const _Float16* W2t = ws + OFF_W2T;
  const _Float16* W3h = ws + OFF_W3H;
  const _Float16* W3t = ws + OFF_W3T;
  const _Float16* W4h = ws + OFF_W4H;

  __shared__ __attribute__((aligned(32))) _Float16 GA[64 * NH];   // 32 KB
  __shared__ __attribute__((aligned(32))) _Float16 GB[64 * NH];   // 32 KB
  __shared__ __attribute__((aligned(32))) _Float16 S1[BM * NH];   // 16 KB
  __shared__ __attribute__((aligned(32))) _Float16 S2[BM * NH];
  __shared__ __attribute__((aligned(32))) _Float16 S3[BM * NH];
  __shared__ __attribute__((aligned(32))) _Float16 Seedbuf[2 * NH];
  __shared__ __attribute__((aligned(32))) _Float16 Xbuf[BM * 32];
  __shared__ __attribute__((aligned(16))) float Jbuf[4 * BM * NST];  // 8 KB
  __shared__ __attribute__((aligned(16))) float Dbuf[BM * NST];      // 2 KB

  _Float16* H1 = GA;            // 32x256, alias (dead before backward)
  _Float16* H2 = GA + BM * NH;  // 32x256
  _Float16* H3 = GB;            // 32x256, alias

  const int tid = threadIdx.x;
  const int lane = tid & 31;
  const int wave = tid >> 5;
  const int row0 = blockIdx.x * BM;
  const int ml = lane & 15;
  const int mb = (lane & 16) ? 8 : 0;

  // ---- stage X (pad K 16->32) ----
  for (int idx = tid; idx < BM * 32; idx += THREADS) {
    int r = idx >> 5, c = idx & 31;
    Xbuf[idx] = (c < NST) ? (_Float16)x[(size_t)(row0 + r) * NST + c]
                          : (_Float16)0.f;
  }
  __syncthreads();

  // ---- layer 1: u1 = X @ W1^T ; h1 = ls(u1), s1 = sig(-u1) ----
#pragma unroll
  for (int t = 0; t < 2; ++t) {
    int nt = wave * 2 + t;
    v16h bf = frag_b_gbl(W1h, 32, nt * 16, 0, lane);
#pragma unroll
    for (int mt = 0; mt < 2; ++mt) {
      v8f acc = {0.f, 0.f, 0.f, 0.f, 0.f, 0.f, 0.f, 0.f};
      v16h af = frag_a_lds(Xbuf, 32, mt * 16 + ml, 0, lane);
      acc = wmma_f16(af, bf, acc);
      int n = nt * 16 + ml;
#pragma unroll
      for (int v = 0; v < 8; ++v) {
        int m = mt * 16 + mb + v;
        float ls, sn;
        act_pair(acc[v], ls, sn);
        H1[m * NH + n] = (_Float16)ls;
        S1[m * NH + n] = (_Float16)sn;
      }
    }
  }
  __syncthreads();

  // ---- layers 2,3: u = Hin @ W^T ; Hout = Hin + ls(u), S = sig(-u) ----
#pragma unroll 1
  for (int layer = 0; layer < 2; ++layer) {
    const _Float16* Hin = layer ? H2 : H1;
    _Float16* Hout = layer ? H3 : H2;
    _Float16* Sout = layer ? S3 : S2;
    const _Float16* Bw = layer ? W3h : W2h;
#pragma unroll
    for (int t = 0; t < 2; ++t) {
      int nt = wave * 2 + t;
      v16h bf[8];
#pragma unroll
      for (int ks = 0; ks < 8; ++ks)
        bf[ks] = frag_b_gbl(Bw, NH, nt * 16, ks * 32, lane);
#pragma unroll
      for (int mt = 0; mt < 2; ++mt) {
        v8f acc = {0.f, 0.f, 0.f, 0.f, 0.f, 0.f, 0.f, 0.f};
#pragma unroll
        for (int ks = 0; ks < 8; ++ks) {
          v16h af = frag_a_lds(Hin, NH, mt * 16 + ml, ks * 32, lane);
          acc = wmma_f16(af, bf[ks], acc);
        }
        int n = nt * 16 + ml;
#pragma unroll
        for (int v = 0; v < 8; ++v) {
          int m = mt * 16 + mb + v;
          float ls, sn;
          act_pair(acc[v], ls, sn);
          Hout[m * NH + n] = (_Float16)((float)Hin[m * NH + n] + ls);
          Sout[m * NH + n] = (_Float16)sn;
        }
      }
    }
    __syncthreads();
  }

  // ---- dstates = H3 @ W4[0:16]^T ----
  if (wave < 2) {
    v16h bf[8];
#pragma unroll
    for (int ks = 0; ks < 8; ++ks) bf[ks] = frag_b_gbl(W4h, NH, 0, ks * 32, lane);
    v8f acc = {0.f, 0.f, 0.f, 0.f, 0.f, 0.f, 0.f, 0.f};
#pragma unroll
    for (int ks = 0; ks < 8; ++ks) {
      v16h af = frag_a_lds(H3, NH, wave * 16 + ml, ks * 32, lane);
      acc = wmma_f16(af, bf[ks], acc);
    }
#pragma unroll
    for (int v = 0; v < 8; ++v)
      Dbuf[(wave * 16 + mb + v) * NST + ml] = acc[v];
  }
  __syncthreads();

  // ---- backward: Jacobian rows for com seeds, 2 seeds per group ----
#pragma unroll 1
  for (int jg = 0; jg < 2; ++jg) {
    // b0: GA[r][k] = W4[16+j][k] * S3[i][k]   (r = jl*BM + i, j = jg*2+jl)
    for (int idx = tid; idx < 2 * BM * NH; idx += THREADS) {
      int r = idx >> 8, k = idx & 255;
      int jl = r >> 5, i = r & 31;
      float sd = (float)W4h[(NST + jg * 2 + jl) * NH + k];
      GA[idx] = (_Float16)(sd * (float)S3[i * NH + k]);
    }
    for (int k = tid; k < 2 * NH; k += THREADS)
      Seedbuf[k] = W4h[(NST + jg * 2) * NH + k];
    __syncthreads();

    // b1: T = GA @ W3 ; GB = seed + T   (g_h2)
#pragma unroll
    for (int t = 0; t < 2; ++t) {
      int nt = wave * 2 + t;
      v16h bf[8];
#pragma unroll
      for (int ks = 0; ks < 8; ++ks)
        bf[ks] = frag_b_gbl(W3t, NH, nt * 16, ks * 32, lane);
#pragma unroll
      for (int mt = 0; mt < 4; ++mt) {
        v8f acc = {0.f, 0.f, 0.f, 0.f, 0.f, 0.f, 0.f, 0.f};
#pragma unroll
        for (int ks = 0; ks < 8; ++ks) {
          v16h af = frag_a_lds(GA, NH, mt * 16 + ml, ks * 32, lane);
          acc = wmma_f16(af, bf[ks], acc);
        }
        int n = nt * 16 + ml;
#pragma unroll
        for (int v = 0; v < 8; ++v) {
          int r = mt * 16 + mb + v;
          int jl = r >> 5;
          GB[r * NH + n] = (_Float16)((float)Seedbuf[jl * NH + n] + acc[v]);
        }
      }
    }
    __syncthreads();

    // b2: U = (GB .* S2) @ W2 ; GA = GB + U   (g_h1)
#pragma unroll
    for (int t = 0; t < 2; ++t) {
      int nt = wave * 2 + t;
      v16h bf[8];
#pragma unroll
      for (int ks = 0; ks < 8; ++ks)
        bf[ks] = frag_b_gbl(W2t, NH, nt * 16, ks * 32, lane);
#pragma unroll
      for (int mt = 0; mt < 4; ++mt) {
        int rA = mt * 16 + ml;
        v8f acc = {0.f, 0.f, 0.f, 0.f, 0.f, 0.f, 0.f, 0.f};
#pragma unroll
        for (int ks = 0; ks < 8; ++ks) {
          v16h g = frag_a_lds(GB, NH, rA, ks * 32, lane);
          v16h s = frag_a_lds(S2, NH, rA & 31, ks * 32, lane);
          acc = wmma_f16(g * s, bf[ks], acc);
        }
        int n = nt * 16 + ml;
#pragma unroll
        for (int v = 0; v < 8; ++v) {
          int r = mt * 16 + mb + v;
          GA[r * NH + n] = (_Float16)((float)GB[r * NH + n] + acc[v]);
        }
      }
    }
    __syncthreads();

    // b3: J = (GA .* S1) @ W1   (N = 16, one tile per wave)
    if (wave < 4) {
      v16h bf[8];
#pragma unroll
      for (int ks = 0; ks < 8; ++ks)
        bf[ks] = frag_b_gbl(W1t, NH, 0, ks * 32, lane);
      int rA = wave * 16 + ml;
      v8f acc = {0.f, 0.f, 0.f, 0.f, 0.f, 0.f, 0.f, 0.f};
#pragma unroll
      for (int ks = 0; ks < 8; ++ks) {
        v16h g = frag_a_lds(GA, NH, rA, ks * 32, lane);
        v16h s = frag_a_lds(S1, NH, rA & 31, ks * 32, lane);
        acc = wmma_f16(g * s, bf[ks], acc);
      }
#pragma unroll
      for (int v = 0; v < 8; ++v) {
        int r = wave * 16 + mb + v;
        int j = jg * 2 + (r >> 5);
        int i = r & 31;
        Jbuf[(j * BM + i) * NST + ml] = acc[v];
      }
    }
    __syncthreads();
  }

  // ---- projection: out = dstates - proj_{span(J0..J3)} dstates (MGS, fp32) ----
  if (tid < BM) {
    float c[4][16], d[16];
#pragma unroll
    for (int j = 0; j < 4; ++j)
#pragma unroll
      for (int k = 0; k < 16; ++k) c[j][k] = Jbuf[(j * BM + tid) * NST + k];
#pragma unroll
    for (int k = 0; k < 16; ++k) d[k] = Dbuf[tid * NST + k];
#pragma unroll 1
    for (int a = 0; a < 4; ++a) {
      for (int b = 0; b < a; ++b) {
        float dp = 0.f;
#pragma unroll
        for (int k = 0; k < 16; ++k) dp += c[a][k] * c[b][k];
#pragma unroll
        for (int k = 0; k < 16; ++k) c[a][k] -= dp * c[b][k];
      }
      float nn = 0.f;
#pragma unroll
      for (int k = 0; k < 16; ++k) nn += c[a][k] * c[a][k];
      if (nn > 1e-20f) {
        float inv = rsqrtf(nn);
#pragma unroll
        for (int k = 0; k < 16; ++k) c[a][k] *= inv;
        float dp = 0.f;
#pragma unroll
        for (int k = 0; k < 16; ++k) dp += d[k] * c[a][k];
#pragma unroll
        for (int k = 0; k < 16; ++k) d[k] -= dp * c[a][k];
      }
    }
#pragma unroll
    for (int k = 0; k < 16; ++k)
      out[(size_t)(row0 + tid) * NST + k] = d[k];
  }
}

// ---------------------------------------------------------------------------
extern "C" void kernel_launch(void* const* d_in, const int* in_sizes, int n_in,
                              void* d_out, int out_size, void* d_ws,
                              size_t ws_size, hipStream_t stream) {
  (void)in_sizes; (void)n_in; (void)out_size; (void)ws_size;
  const float* x  = (const float*)d_in[0];
  const float* W1 = (const float*)d_in[1];
  const float* W2 = (const float*)d_in[2];
  const float* W3 = (const float*)d_in[3];
  const float* W4 = (const float*)d_in[4];
  _Float16* ws = (_Float16*)d_ws;
  float* out = (float*)d_out;

  CoMet_convert_weights<<<256, 256, 0, stream>>>(W1, W2, W3, W4, ws);
  CoMet_fused<<<BATCH / BM, THREADS, 0, stream>>>(x, ws, out);
}
